// QLlamaAttention_67190468379218
// MI455X (gfx1250) — matrix operand order repaired and checked by reading, not compiled, not run on hardware
//
#include <hip/hip_runtime.h>
#include <hip/hip_bf16.h>

typedef _Float16 v16h __attribute__((ext_vector_type(16)));
typedef _Float16 h8   __attribute__((ext_vector_type(8)));
typedef float    v8f  __attribute__((ext_vector_type(8)));

#define WMMA_F16(a,b,c) __builtin_amdgcn_wmma_f32_16x16x32_f16(false,(a),false,(b),(short)0,(c),false,false)

static __device__ __forceinline__ v16h mk16(h8 lo, h8 hi) {
  return __builtin_shufflevector(lo, hi, 0,1,2,3,4,5,6,7,8,9,10,11,12,13,14,15);
}
static __device__ __forceinline__ v8f zero8() {
  v8f z = {0.f,0.f,0.f,0.f,0.f,0.f,0.f,0.f};
  return z;
}

// HIDDEN=4096, NH=32, HD=128, B=2, S=2048, T=B*S=4096
// quant tiers: cols [0,2560) 4-bit (qmax 7), [2560,3584) 6-bit (qmax 31), [3584,4096) 8-bit (qmax 127)

// ---------------- mixed fake-quant (+optional gather) -> f16 ----------------
template <typename T>
__global__ __launch_bounds__(256) void quant_kernel(const T* __restrict__ X,
                                                    const int* __restrict__ reorder,
                                                    _Float16* __restrict__ out) {
  const int gid  = blockIdx.x * 256 + threadIdx.x;   // 4096 rows * 128 blocks * 32 lanes
  const int gw   = gid >> 5;
  const int lane = gid & 31;
  const int row  = gw >> 7;
  const int blk  = gw & 127;
  const int c    = blk * 32 + lane;
  const int src  = reorder ? reorder[c] : c;
  float v = (float)X[(size_t)row * 4096 + src];
  const float qmax = (c < 2560) ? 7.f : ((c < 3584) ? 31.f : 127.f);
  float a = fabsf(v);
  a = fmaxf(a, __shfl_xor(a, 1));
  a = fmaxf(a, __shfl_xor(a, 2));
  a = fmaxf(a, __shfl_xor(a, 4));
  a = fmaxf(a, __shfl_xor(a, 8));
  a = fmaxf(a, __shfl_xor(a, 16));
  const float scale = fmaxf(a / qmax, 1e-8f);
  float q = rintf(v / scale);
  q = fminf(fmaxf(q, -qmax), qmax);
  out[(size_t)row * 4096 + c] = (_Float16)(q * scale);
}

// ---------------- C[M,N] = A[M,K] * B[N,K]^T  (f16 in, f32 acc) ----------------
// block tile 128x256, 8 waves, 64x64 per wave, double-buffered LDS, 1 barrier/K-step
template <bool OUTF32>
__global__ __launch_bounds__(256) void gemm_nt(const _Float16* __restrict__ A,
                                               const _Float16* __restrict__ B,
                                               void* __restrict__ C) {
  __shared__ __align__(16) _Float16 As[2 * 128 * 40];
  __shared__ __align__(16) _Float16 Bs[2 * 256 * 40];
  const int tid = threadIdx.x;
  const int w = tid >> 5, lane = tid & 31;
  const int hl = lane >> 4, lr = lane & 15;
  const int m0 = blockIdx.y * 128, n0 = blockIdx.x * 256;
  const int wm = (w >> 2) * 64, wn = (w & 3) * 64;

  // staging: A -> 16 f16/thread (rows tid>>1), B -> 32 f16/thread (row tid)
  const _Float16* Ag = A + (size_t)(m0 + (tid >> 1)) * 4096 + (tid & 1) * 16;
  const _Float16* Bg = B + (size_t)(n0 + tid) * 4096;
  _Float16* Asw = As + (tid >> 1) * 40 + (tid & 1) * 16;
  _Float16* Bsw = Bs + tid * 40;

  v8f acc[4][4];
#pragma unroll
  for (int i = 0; i < 4; ++i)
#pragma unroll
    for (int j = 0; j < 4; ++j) acc[i][j] = zero8();

  h8 ar0, ar1, br0, br1, br2, br3;
  // prime k0 = 0
  ar0 = *(const h8*)(Ag);
  ar1 = *(const h8*)(Ag + 8);
  br0 = *(const h8*)(Bg);
  br1 = *(const h8*)(Bg + 8);
  br2 = *(const h8*)(Bg + 16);
  br3 = *(const h8*)(Bg + 24);
  *(h8*)(Asw)      = ar0;
  *(h8*)(Asw + 8)  = ar1;
  *(h8*)(Bsw)      = br0;
  *(h8*)(Bsw + 8)  = br1;
  *(h8*)(Bsw + 16) = br2;
  *(h8*)(Bsw + 24) = br3;
  __syncthreads();

  int buf = 0;
  for (int k0 = 0; k0 < 4096; k0 += 32, buf ^= 1) {
    const bool more = (k0 + 32) < 4096;
    if (more) {  // issue next-step global loads before compute (latency hiding)
      ar0 = *(const h8*)(Ag + k0 + 32);
      ar1 = *(const h8*)(Ag + k0 + 40);
      br0 = *(const h8*)(Bg + k0 + 32);
      br1 = *(const h8*)(Bg + k0 + 40);
      br2 = *(const h8*)(Bg + k0 + 48);
      br3 = *(const h8*)(Bg + k0 + 56);
      if (k0 + 512 < 4096) {
        __builtin_prefetch(Ag + k0 + 512, 0, 0);
        __builtin_prefetch(Bg + k0 + 512, 0, 0);
      }
    }

    const _Float16* Ab = As + buf * (128 * 40);
    const _Float16* Bb = Bs + buf * (256 * 40);
    v16h af[4];
#pragma unroll
    for (int i = 0; i < 4; ++i) {
      const _Float16* p = Ab + (wm + 16 * i + lr) * 40;
      af[i] = mk16(*(const h8*)(p + hl * 16), *(const h8*)(p + hl * 16 + 8));
    }
#pragma unroll
    for (int j = 0; j < 4; ++j) {
      const _Float16* p = Bb + (wn + 16 * j + lr) * 40 + hl * 16;
      v16h bf = mk16(*(const h8*)(p), *(const h8*)(p + 8));
#pragma unroll
      for (int i = 0; i < 4; ++i)
        acc[i][j] = WMMA_F16(af[i], bf, acc[i][j]);
    }

    if (more) {  // stage next buffer; readers of it were fenced by previous barrier
      _Float16* Aw = Asw + (buf ^ 1) * (128 * 40);
      _Float16* Bw = Bsw + (buf ^ 1) * (256 * 40);
      *(h8*)(Aw)      = ar0;
      *(h8*)(Aw + 8)  = ar1;
      *(h8*)(Bw)      = br0;
      *(h8*)(Bw + 8)  = br1;
      *(h8*)(Bw + 16) = br2;
      *(h8*)(Bw + 24) = br3;
    }
    __syncthreads();
  }

#pragma unroll
  for (int i = 0; i < 4; ++i)
#pragma unroll
    for (int j = 0; j < 4; ++j)
#pragma unroll
      for (int r = 0; r < 8; ++r) {
        const int mg = m0 + wm + 16 * i + 8 * hl + r;
        const int ng = n0 + wn + 16 * j + lr;
        const size_t idx = (size_t)mg * 4096 + ng;
        if constexpr (OUTF32) ((float*)C)[idx] = acc[i][j][r];
        else                  ((_Float16*)C)[idx] = (_Float16)acc[i][j][r];
      }
}

// ---------------- RoPE in place on f16 [T, 4096] ----------------
__global__ __launch_bounds__(256) void rope_kernel(_Float16* __restrict__ X,
                                                   const int* __restrict__ pos_ids) {
  const int idx = blockIdx.x * 256 + threadIdx.x;  // T*32*64 = 8388608
  const int i = idx & 63;
  const int h = (idx >> 6) & 31;
  const int t = idx >> 11;
  const float pos = (float)pos_ids[t];
  const float ang = pos * __powf(10000.f, -(float)i * (1.f / 64.f));
  float s, c;
  __sincosf(ang, &s, &c);
  const size_t base = (size_t)t * 4096 + h * 128 + i;
  const float q1 = (float)X[base];
  const float q2 = (float)X[base + 64];
  X[base]      = (_Float16)(q1 * c - q2 * s);
  X[base + 64] = (_Float16)(q2 * c + q1 * s);
}

// ---------------- causal flash attention: Q,K,V f16 [T,4096] -> O f16 [T,4096] ----------------
__global__ __launch_bounds__(256) void flash_kernel(const _Float16* __restrict__ Qf,
                                                    const _Float16* __restrict__ Kf,
                                                    const _Float16* __restrict__ Vf,
                                                    _Float16* __restrict__ Of) {
  __shared__ __align__(16) _Float16 Ks[32 * 136];     // K block [32 keys][128 d]
  __shared__ __align__(16) _Float16 Vt[128 * 40];     // V block transposed [128 d][32 keys]
  __shared__ __align__(16) _Float16 Ps[8 * 16 * 40];  // per-wave P tile [16 q][32 keys]
  const int tid = threadIdx.x;
  const int w = tid >> 5, lane = tid & 31;
  const int hl = lane >> 4, lr = lane & 15;
  const int q0 = blockIdx.x * 128;
  const int bh = blockIdx.y;
  const int b = bh >> 5, h = bh & 31;
  const int wrow = q0 + w * 16;

  // preload Q fragments (16 q rows x 128 d = 4 A-frags of 16x32)
  v16h aq[4];
  const size_t rowQ = (size_t)(b * 2048 + wrow + lr) * 4096 + h * 128;
#pragma unroll
  for (int kc = 0; kc < 4; ++kc) {
    const _Float16* p = Qf + rowQ + kc * 32;
    aq[kc] = mk16(*(const h8*)(p + hl * 16), *(const h8*)(p + hl * 16 + 8));
  }

  v8f o[8];
#pragma unroll
  for (int j = 0; j < 8; ++j) o[j] = zero8();
  float mrow[8], lrow[8];
#pragma unroll
  for (int r = 0; r < 8; ++r) { mrow[r] = -1e30f; lrow[r] = 0.f; }

  const int kbmax = q0 / 32 + 4;   // causal: last key block touching q0+127
  const int wmax = wrow + 15;
  const int ldn = tid >> 3;        // 0..31 key row for loads
  const int ldc = (tid & 7) * 16;  // 0..112 d col for loads
  const float SC = 0.08838834764831845f;  // 1/sqrt(128)

  for (int kb = 0; kb < kbmax; ++kb) {
    __syncthreads();
    {
      const size_t rg = (size_t)(b * 2048 + kb * 32 + ldn) * 4096 + h * 128 + ldc;
      h8 klo = *(const h8*)(Kf + rg);
      h8 khi = *(const h8*)(Kf + rg + 8);
      *(h8*)(&Ks[ldn * 136 + ldc])     = klo;
      *(h8*)(&Ks[ldn * 136 + ldc + 8]) = khi;
      h8 vlo = *(const h8*)(Vf + rg);
      h8 vhi = *(const h8*)(Vf + rg + 8);
#pragma unroll
      for (int e = 0; e < 8; ++e) {
        Vt[(ldc + e) * 40 + ldn]     = vlo[e];
        Vt[(ldc + 8 + e) * 40 + ldn] = vhi[e];
      }
    }
    __syncthreads();
    if (kb * 32 > wmax) continue;  // wave fully masked (uniform per wave)

    // S = Q * K^T : 16 x 32 scores in two 16x16 C tiles
    v8f s0 = zero8(), s1 = zero8();
#pragma unroll
    for (int kc = 0; kc < 4; ++kc) {
      const _Float16* p0 = &Ks[lr * 136 + kc * 32 + hl * 16];
      const _Float16* p1 = &Ks[(16 + lr) * 136 + kc * 32 + hl * 16];
      v16h b0 = mk16(*(const h8*)p0, *(const h8*)(p0 + 8));
      v16h b1 = mk16(*(const h8*)p1, *(const h8*)(p1 + 8));
      s0 = WMMA_F16(aq[kc], b0, s0);
      s1 = WMMA_F16(aq[kc], b1, s1);
    }

    // online softmax (row = 8*hl + r, col = 16*j + lr)
    float al[8];
#pragma unroll
    for (int r = 0; r < 8; ++r) {
      float v0 = s0[r] * SC, v1 = s1[r] * SC;
      const int mg  = wrow + 8 * hl + r;
      const int n0g = kb * 32 + lr;
      if (n0g > mg)      v0 = -1e30f;
      if (n0g + 16 > mg) v1 = -1e30f;
      float rm = fmaxf(v0, v1);
      rm = fmaxf(rm, __shfl_xor(rm, 1));
      rm = fmaxf(rm, __shfl_xor(rm, 2));
      rm = fmaxf(rm, __shfl_xor(rm, 4));
      rm = fmaxf(rm, __shfl_xor(rm, 8));
      const float mn = fmaxf(mrow[r], rm);
      const float a  = __expf(mrow[r] - mn);
      const float p0v = __expf(v0 - mn);
      const float p1v = __expf(v1 - mn);
      float ps = p0v + p1v;
      ps += __shfl_xor(ps, 1);
      ps += __shfl_xor(ps, 2);
      ps += __shfl_xor(ps, 4);
      ps += __shfl_xor(ps, 8);
      lrow[r] = lrow[r] * a + ps;
      mrow[r] = mn;
      al[r] = a;
      _Float16* pp = &Ps[w * 640 + (8 * hl + r) * 40];
      pp[lr]      = (_Float16)p0v;
      pp[16 + lr] = (_Float16)p1v;
    }
#pragma unroll
    for (int j = 0; j < 8; ++j)
#pragma unroll
      for (int r = 0; r < 8; ++r) o[j][r] = o[j][r] * al[r];

    // O += P * V  (A = P 16x32 from wave-private LDS, B = Vt contiguous)
    const _Float16* pa = &Ps[w * 640 + lr * 40];
    v16h ap = mk16(*(const h8*)(pa + hl * 16), *(const h8*)(pa + hl * 16 + 8));
#pragma unroll
    for (int j = 0; j < 8; ++j) {
      const _Float16* pv = &Vt[(j * 16 + lr) * 40 + hl * 16];
      v16h bv = mk16(*(const h8*)pv, *(const h8*)(pv + 8));
      o[j] = WMMA_F16(ap, bv, o[j]);
    }
  }

  // normalize + store f16
#pragma unroll
  for (int j = 0; j < 8; ++j)
#pragma unroll
    for (int r = 0; r < 8; ++r) {
      const int mg = wrow + 8 * hl + r;
      const size_t idx = (size_t)(b * 2048 + mg) * 4096 + h * 128 + j * 16 + lr;
      Of[idx] = (_Float16)(o[j][r] / lrow[r]);
    }
}

// ---------------- host launch ----------------
extern "C" void kernel_launch(void* const* d_in, const int* in_sizes, int n_in,
                              void* d_out, int out_size, void* d_ws, size_t ws_size,
                              hipStream_t stream) {
  const float* hidden = (const float*)d_in[0];
  const float* Wq = (const float*)d_in[1];
  const float* Wk = (const float*)d_in[2];
  const float* Wv = (const float*)d_in[3];
  const float* Wo = (const float*)d_in[4];
  const int* pos = (const int*)d_in[5];
  const int* reorder = (const int*)d_in[6];
  float* out = (float*)d_out;

  char* ws = (char*)d_ws;
  const size_t SEG = (size_t)4096 * 4096 * sizeof(_Float16);  // 32 MB per f16 matrix
  _Float16* Xq  = (_Float16*)(ws + 0 * SEG);
  _Float16* Wqq = (_Float16*)(ws + 1 * SEG);
  _Float16* Wkq = (_Float16*)(ws + 2 * SEG);
  _Float16* Wvq = (_Float16*)(ws + 3 * SEG);
  _Float16* Woq = (_Float16*)(ws + 4 * SEG);
  _Float16* Qh  = (_Float16*)(ws + 5 * SEG);
  _Float16* Kh  = (_Float16*)(ws + 6 * SEG);
  _Float16* Vh  = (_Float16*)(ws + 7 * SEG);
  _Float16* Ao  = (_Float16*)(ws + 8 * SEG);
  _Float16* Aq  = Xq;  // reuse after QKV GEMMs

  const dim3 qblk(256), qgrid(65536);     // 4096 rows * 128 blocks, one wave each
  quant_kernel<float><<<qgrid, qblk, 0, stream>>>(hidden, reorder, Xq);
  quant_kernel<float><<<qgrid, qblk, 0, stream>>>(Wq, nullptr, Wqq);
  quant_kernel<float><<<qgrid, qblk, 0, stream>>>(Wk, nullptr, Wkq);
  quant_kernel<float><<<qgrid, qblk, 0, stream>>>(Wv, nullptr, Wvq);
  quant_kernel<float><<<qgrid, qblk, 0, stream>>>(Wo, nullptr, Woq);

  const dim3 ggrid(16, 32), gblk(256);    // 256-wide x 128-tall C tiles
  gemm_nt<false><<<ggrid, gblk, 0, stream>>>(Xq, Wqq, (void*)Qh);
  gemm_nt<false><<<ggrid, gblk, 0, stream>>>(Xq, Wkq, (void*)Kh);
  gemm_nt<false><<<ggrid, gblk, 0, stream>>>(Xq, Wvq, (void*)Vh);

  rope_kernel<<<32768, 256, 0, stream>>>(Qh, pos);
  rope_kernel<<<32768, 256, 0, stream>>>(Kh, pos);

  flash_kernel<<<dim3(16, 64), 256, 0, stream>>>(Qh, Kh, Vh, Ao);

  quant_kernel<_Float16><<<qgrid, qblk, 0, stream>>>(Ao, reorder, Aq);
  gemm_nt<true><<<ggrid, gblk, 0, stream>>>(Aq, Woq, (void*)out);
}
